// BathymetricGNN_11450382811841
// MI455X (gfx1250) — matrix operand order repaired and checked by reading, not compiled, not run on hardware
//
#include <hip/hip_runtime.h>
#include <hip/hip_bf16.h>
#include <stdint.h>

// ---------------------------------------------------------------------------
// BathymetricGNN forward for MI455X (gfx1250, wave32, WMMA).
//   - Dense node transforms (h = x @ W)  -> v_wmma_f32_16x16x32_bf16 with
//     pre-converted bf16 operands laid out so every fragment = 2x b128 loads.
//   - Edge softmax/aggregation           -> coalesced fp32 atomics (BW bound)
//   - BatchNorm (batch stats, biased var), GAT bias folded away by BN.
// Outputs (flat, float): logits[N*3] | probs[N*3] | pred[N] | conf[N] | corr[N]
// ---------------------------------------------------------------------------

#define HH     64
#define EPSBN  1e-5f

typedef __bf16 v16bf __attribute__((ext_vector_type(16)));
typedef __bf16 v8bf  __attribute__((ext_vector_type(8)));
typedef float  v8f   __attribute__((ext_vector_type(8)));

__device__ __forceinline__ uint32_t f2bf_bits(float f) {
    // round-to-nearest-even fp32 -> bf16 bit pattern
    union { float f; uint32_t u; } x; x.f = f;
    return (x.u + 0x7FFFu + ((x.u >> 16) & 1u)) >> 16;
}

__device__ __forceinline__ void atomicMaxF(float* addr, float val) {
    if (val >= 0.0f) atomicMax((int*)addr, __float_as_int(val));
    else             atomicMin((unsigned int*)addr, __float_as_uint(val));
}

// fp32 -> bf16 row-major streaming convert (2 elements / thread, packed store)
__global__ void k_cvt_bf16(const float* __restrict__ in, uint32_t* __restrict__ out,
                           long long npairs) {
    const long long i = (long long)blockIdx.x * blockDim.x + threadIdx.x;
    if (i >= npairs) return;
    const float2 f = ((const float2*)in)[i];
    out[i] = f2bf_bits(f.x) | (f2bf_bits(f.y) << 16);
}

// W[K,O] fp32 -> Wt[O,K] bf16 (transpose-convert; weights are tiny/L2-resident)
__global__ void k_cvt_w(const float* __restrict__ W, uint16_t* __restrict__ Wt,
                        int K, int O) {
    const int i = blockIdx.x * blockDim.x + threadIdx.x;
    if (i >= K * O) return;
    const int o = i / K, k = i % K;
    Wt[(size_t)o * K + k] = (uint16_t)f2bf_bits(W[(size_t)k * O + o]);
}

// ---------------------------------------------------------------------------
// WMMA GEMM: C[Nr,O] = A[Nr,K] @ B[K,O], bf16 operands, fp32 accumulate.
//   A  : bf16 row-major [Nr,K]
//   Bt : bf16 col-major of B, i.e. row-major [O,K]
// One wave computes a 16x64 output strip (4 accumulators), K-step 32.
// Fragment layout: contiguous k = 16*(lane/16) + s for A and B (wave32); every fragment is loaded
// as two contiguous 16-byte chunks (global_load_b128).
// ---------------------------------------------------------------------------
__global__ void k_gemm_bf16(const __bf16* __restrict__ A, const __bf16* __restrict__ Bt,
                            float* __restrict__ C, int Nr, int K, int O) {
    const int lane = threadIdx.x & 31;
    const int wave = blockIdx.x * (blockDim.x >> 5) + (threadIdx.x >> 5);
    const int colGroups = O >> 6;                 // 64 output cols per wave
    const int rowTiles  = (Nr + 15) >> 4;
    if (wave >= rowTiles * colGroups) return;
    const int rowBase = (wave / colGroups) << 4;
    const int colBase = (wave % colGroups) << 6;

    v8f acc0 = {}, acc1 = {}, acc2 = {}, acc3 = {};

    int m = rowBase + (lane & 15);
    if (m >= Nr) m = Nr - 1;                      // safe over-read row
    const int ka = (lane >> 4) << 4;              // A: K half-offset (0 / 16)
    const int kb = (lane >> 4) << 4;              // B: K half-offset (0 / 16)
    const int nb = lane & 15;

    const __bf16* arow = A + (size_t)m * K + ka;
    const __bf16* bp0 = Bt + (size_t)(colBase + nb)      * K + kb;
    const __bf16* bp1 = Bt + (size_t)(colBase + 16 + nb) * K + kb;
    const __bf16* bp2 = Bt + (size_t)(colBase + 32 + nb) * K + kb;
    const __bf16* bp3 = Bt + (size_t)(colBase + 48 + nb) * K + kb;

    for (int k0 = 0; k0 < K; k0 += 32) {
        // ---- A fragment: lane holds k = ka..ka+15 (contiguous) ----
        const v8bf alo = *(const v8bf*)(arow + k0);
        const v8bf ahi = *(const v8bf*)(arow + k0 + 8);
        const v16bf a = __builtin_shufflevector(alo, ahi,
            0, 1, 2, 3, 4, 5, 6, 7, 8, 9, 10, 11, 12, 13, 14, 15);

        // ---- 4x B fragment: lane holds col n, k = kb..kb+15 (contiguous) ----
        v8bf blo, bhi;
        v16bf b;

        blo = *(const v8bf*)(bp0 + k0); bhi = *(const v8bf*)(bp0 + k0 + 8);
        b = __builtin_shufflevector(blo, bhi, 0,1,2,3,4,5,6,7,8,9,10,11,12,13,14,15);
        acc0 = __builtin_amdgcn_wmma_f32_16x16x32_bf16(false, a, false, b, (short)0, acc0, false, false);

        blo = *(const v8bf*)(bp1 + k0); bhi = *(const v8bf*)(bp1 + k0 + 8);
        b = __builtin_shufflevector(blo, bhi, 0,1,2,3,4,5,6,7,8,9,10,11,12,13,14,15);
        acc1 = __builtin_amdgcn_wmma_f32_16x16x32_bf16(false, a, false, b, (short)0, acc1, false, false);

        blo = *(const v8bf*)(bp2 + k0); bhi = *(const v8bf*)(bp2 + k0 + 8);
        b = __builtin_shufflevector(blo, bhi, 0,1,2,3,4,5,6,7,8,9,10,11,12,13,14,15);
        acc2 = __builtin_amdgcn_wmma_f32_16x16x32_bf16(false, a, false, b, (short)0, acc2, false, false);

        blo = *(const v8bf*)(bp3 + k0); bhi = *(const v8bf*)(bp3 + k0 + 8);
        b = __builtin_shufflevector(blo, bhi, 0,1,2,3,4,5,6,7,8,9,10,11,12,13,14,15);
        acc3 = __builtin_amdgcn_wmma_f32_16x16x32_bf16(false, a, false, b, (short)0, acc3, false, false);
    }

    // ---- store D: VGPR r -> row rowBase + 8*(lane>=16) + r ----
    const int rbase = rowBase + ((lane >> 4) << 3);
#pragma unroll
    for (int r = 0; r < 8; ++r) {
        const int row = rbase + r;
        if (row < Nr) {
            float* crow = C + (size_t)row * O + colBase + nb;
            crow[0]  = acc0[r];
            crow[16] = acc1[r];
            crow[32] = acc2[r];
            crow[48] = acc3[r];
        }
    }
}

// ---------------------------------------------------------------------------
// Feature extractor: h0 = relu(x @ W1 + b1) @ W2 + b2   (12->64->64)
// 64 threads per node, 4 nodes per 256-thread block, staged through LDS.
// ---------------------------------------------------------------------------
__global__ void k_fe(const float* __restrict__ x, const float* __restrict__ w1,
                     const float* __restrict__ b1, const float* __restrict__ w2,
                     const float* __restrict__ b2, float* __restrict__ h0, int N) {
    __shared__ float sx[4][12];
    __shared__ float sh[4][64];
    const int slot = threadIdx.x >> 6;
    const int j    = threadIdx.x & 63;
    const int n    = blockIdx.x * 4 + slot;
    if (n < N && j < 12) sx[slot][j] = x[(size_t)n * 12 + j];
    __syncthreads();
    if (n < N) {
        float acc = b1[j];
#pragma unroll
        for (int k = 0; k < 12; ++k) acc += sx[slot][k] * w1[k * 64 + j];
        sh[slot][j] = fmaxf(acc, 0.0f);
    }
    __syncthreads();
    if (n < N) {
        float acc = b2[j];
#pragma unroll
        for (int k = 0; k < 64; ++k) acc += sh[slot][k] * w2[k * 64 + j];
        h0[(size_t)n * 64 + j] = acc;
    }
}

// per-(node,head) attention coefficients: asrc = <h, a_s>, adst = <h, a_d>
__global__ void k_attn(const float* __restrict__ hT, const float* __restrict__ a_s,
                       const float* __restrict__ a_d, float* __restrict__ asrc,
                       float* __restrict__ adst, int N, int heads) {
    const int idx = blockIdx.x * blockDim.x + threadIdx.x;
    if (idx >= N * heads) return;
    const int hd = idx % heads, n = idx / heads;
    const float* row = hT + (size_t)n * heads * HH + hd * HH;
    const float* vs = a_s + hd * HH;
    const float* vd = a_d + hd * HH;
    float s = 0.f, d = 0.f;
#pragma unroll
    for (int k = 0; k < HH; ++k) { const float v = row[k]; s += v * vs[k]; d += v * vd[k]; }
    asrc[idx] = s;
    adst[idx] = d;
}

__global__ void k_fill_u32(uint32_t* __restrict__ p, uint32_t v, long long n) {
    const long long i = (long long)blockIdx.x * blockDim.x + threadIdx.x;
    if (i < n) p[i] = v;
}

// alpha = leaky_relu(asrc[s] + adst[d], 0.2); segment-max into amax[d]
__global__ void k_alpha(const int* __restrict__ src, const int* __restrict__ dst,
                        int E, int Etot, const float* __restrict__ asrc,
                        const float* __restrict__ adst, float* __restrict__ alpha,
                        float* __restrict__ amax, int heads) {
    const long long idx = (long long)blockIdx.x * blockDim.x + threadIdx.x;
    if (idx >= (long long)Etot * heads) return;
    const int hd = (int)(idx % heads), e = (int)(idx / heads);
    int s, d;
    if (e < E) { s = src[e]; d = dst[e]; } else { s = e - E; d = s; }   // self loops
    float a = asrc[(size_t)s * heads + hd] + adst[(size_t)d * heads + hd];
    a = (a > 0.0f) ? a : 0.2f * a;
    alpha[idx] = a;
    atomicMaxF(&amax[(size_t)d * heads + hd], a);
}

// ex = exp(alpha - amax[d]); segment-sum into den[d]; ex overwrites alpha
__global__ void k_expsum(const int* __restrict__ src, const int* __restrict__ dst,
                         int E, int Etot, float* __restrict__ alpha,
                         const float* __restrict__ amax, float* __restrict__ den, int heads) {
    const long long idx = (long long)blockIdx.x * blockDim.x + threadIdx.x;
    if (idx >= (long long)Etot * heads) return;
    const int hd = (int)(idx % heads), e = (int)(idx / heads);
    int d;
    if (e < E) { d = dst[e]; } else { d = e - E; }
    const float ex = __expf(alpha[idx] - amax[(size_t)d * heads + hd]);
    alpha[idx] = ex;
    atomicAdd(&den[(size_t)d * heads + hd], ex);
}

// agg[d, hd, :] += hT[s, hd, :] * (ex / den[d,hd]) — one wave per (edge, head)
__global__ void k_agg(const int* __restrict__ src, const int* __restrict__ dst,
                      int E, int Etot, const float* __restrict__ hT,
                      const float* __restrict__ ex, const float* __restrict__ den,
                      float* __restrict__ agg, int heads) {
    const long long t = (long long)blockIdx.x * blockDim.x + threadIdx.x;
    if (t >= (long long)Etot * heads * 32) return;
    const int lane = (int)(t & 31);
    const long long p = t >> 5;
    const int hd = (int)(p % heads), e = (int)(p / heads);
    int s, d;
    if (e < E) { s = src[e]; d = dst[e]; } else { s = e - E; d = s; }
    const float w = ex[(size_t)e * heads + hd] / (den[(size_t)d * heads + hd] + 1e-16f);
    const int D = heads * HH;
    const float* hrow = hT + (size_t)s * D + hd * HH;
    float* orow = agg + (size_t)d * D + hd * HH;
    atomicAdd(&orow[lane],      hrow[lane]      * w);
    atomicAdd(&orow[lane + 32], hrow[lane + 32] * w);
}

// column-wise sum and sum-of-squares (thread t owns column t; coalesced rows)
__global__ void k_bnstats(const float* __restrict__ xin, float* __restrict__ sums,
                          float* __restrict__ sqs, int N, int D) {
    const int c = threadIdx.x;
    if (c >= D) return;
    float s = 0.f, q = 0.f;
    for (int row = blockIdx.x; row < N; row += gridDim.x) {
        const float v = xin[(size_t)row * D + c];
        s += v; q += v * v;
    }
    atomicAdd(&sums[c], s);
    atomicAdd(&sqs[c], q);
}

__global__ void k_bnapply(float* __restrict__ x, const float* __restrict__ sums,
                          const float* __restrict__ sqs, const float* __restrict__ g,
                          const float* __restrict__ be, int N, int D, int relu) {
    const long long idx = (long long)blockIdx.x * blockDim.x + threadIdx.x;
    if (idx >= (long long)N * D) return;
    const int c = (int)(idx % D);
    const float m = sums[c] / (float)N;
    const float v = sqs[c] / (float)N - m * m;
    float y = (x[idx] - m) * rsqrtf(v + EPSBN) * g[c] + be[c];
    if (relu) y = fmaxf(y, 0.0f);
    x[idx] = y;
}

// final MLP heads: cls(64->32->3)+softmax+argmax, conf(->1)+sigmoid, corr(->1)
__global__ void k_heads(const float* __restrict__ h,
                        const float* cw1, const float* cb1, const float* cw2, const float* cb2,
                        const float* fw1, const float* fb1, const float* fw2, const float* fb2,
                        const float* rw1, const float* rb1, const float* rw2, const float* rb2,
                        float* __restrict__ out, int N) {
    __shared__ float sh[4][64];
    __shared__ float hidC[4][32], hidF[4][32], hidR[4][32];
    __shared__ float lg[4][3];
    const int slot = threadIdx.x >> 6;
    const int j    = threadIdx.x & 63;
    const int n    = blockIdx.x * 4 + slot;
    if (n < N) sh[slot][j] = h[(size_t)n * 64 + j];
    __syncthreads();
    if (n < N) {
        if (j < 32) {
            float a = cb1[j], b = rb1[j];
#pragma unroll
            for (int k = 0; k < 64; ++k) {
                const float v = sh[slot][k];
                a += v * cw1[k * 32 + j];
                b += v * rw1[k * 32 + j];
            }
            hidC[slot][j] = fmaxf(a, 0.0f);
            hidR[slot][j] = fmaxf(b, 0.0f);
        } else {
            const int jj = j - 32;
            float a = fb1[jj];
#pragma unroll
            for (int k = 0; k < 64; ++k) a += sh[slot][k] * fw1[k * 32 + jj];
            hidF[slot][jj] = fmaxf(a, 0.0f);
        }
    }
    __syncthreads();
    if (n < N && j < 3) {
        float a = cb2[j];
#pragma unroll
        for (int k = 0; k < 32; ++k) a += hidC[slot][k] * cw2[k * 3 + j];
        lg[slot][j] = a;
    }
    __syncthreads();
    if (n < N && j == 0) {
        const float l0 = lg[slot][0], l1 = lg[slot][1], l2 = lg[slot][2];
        const float mx = fmaxf(l0, fmaxf(l1, l2));
        const float e0 = __expf(l0 - mx), e1 = __expf(l1 - mx), e2 = __expf(l2 - mx);
        const float inv = 1.0f / (e0 + e1 + e2);
        int pred = 0; float best = l0;
        if (l1 > best) { best = l1; pred = 1; }
        if (l2 > best) { best = l2; pred = 2; }
        float cf = fb2[0], cr = rb2[0];
#pragma unroll
        for (int k = 0; k < 32; ++k) { cf += hidF[slot][k] * fw2[k]; cr += hidR[slot][k] * rw2[k]; }
        const size_t N3 = (size_t)N * 3;
        out[(size_t)n * 3 + 0] = l0;
        out[(size_t)n * 3 + 1] = l1;
        out[(size_t)n * 3 + 2] = l2;
        out[N3 + (size_t)n * 3 + 0] = e0 * inv;
        out[N3 + (size_t)n * 3 + 1] = e1 * inv;
        out[N3 + (size_t)n * 3 + 2] = e2 * inv;
        out[2 * N3 + n] = (float)pred;
        out[2 * N3 + N + n] = 1.0f / (1.0f + __expf(-cf));
        out[2 * N3 + 2 * (size_t)N + n] = cr;
    }
}

// ---------------------------------------------------------------------------
// Host-side orchestration (all launches on `stream`; graph-capture safe).
// ---------------------------------------------------------------------------
extern "C" void kernel_launch(void* const* d_in, const int* in_sizes, int n_in,
                              void* d_out, int out_size, void* d_ws, size_t ws_size,
                              hipStream_t stream) {
    (void)n_in; (void)out_size; (void)ws_size;
    const int N = in_sizes[0] / 12;
    const int E = in_sizes[1] / 2;
    const int Etot = E + N;                       // PyG GAT adds self-loops

    const float* x     = (const float*)d_in[0];
    const int*   src   = (const int*)d_in[1];
    const int*   dst   = src + E;
    const float* fe_w1 = (const float*)d_in[2];
    const float* fe_b1 = (const float*)d_in[3];
    const float* fe_w2 = (const float*)d_in[4];
    const float* fe_b2 = (const float*)d_in[5];
    // per-GAT-layer params start at index 6, six tensors each: W, as, ad, b, g, be
    const int pbase[4] = {6, 12, 18, 24};
    const float* cls_w1 = (const float*)d_in[30]; const float* cls_b1 = (const float*)d_in[31];
    const float* cls_w2 = (const float*)d_in[32]; const float* cls_b2 = (const float*)d_in[33];
    const float* cnf_w1 = (const float*)d_in[34]; const float* cnf_b1 = (const float*)d_in[35];
    const float* cnf_w2 = (const float*)d_in[36]; const float* cnf_b2 = (const float*)d_in[37];
    const float* cor_w1 = (const float*)d_in[38]; const float* cor_b1 = (const float*)d_in[39];
    const float* cor_w2 = (const float*)d_in[40]; const float* cor_b2 = (const float*)d_in[41];

    // workspace carve-up (float units), ~140 MB total
    float* ws = (float*)d_ws;
    size_t off = 0;
    auto carve = [&](size_t nflt) { float* p = ws + off; off += (nflt + 63) & ~(size_t)63; return p; };
    float*  bufA  = carve((size_t)N * 256);       // hin / agg (ping), fp32
    float*  bufB  = carve((size_t)N * 256);       // hT = hin @ W (pong), fp32
    __bf16* bufAh = (__bf16*)carve((size_t)N * 128);  // bf16 copy of hin (N*256 bf16)
    uint16_t* Wt  = (uint16_t*)carve(32768);      // bf16 transposed weights (<=256x256)
    float* alpha = carve((size_t)Etot * 4);       // per-(edge,head) alpha -> ex
    float* asrc  = carve((size_t)N * 4);
    float* adst  = carve((size_t)N * 4);
    float* amax  = carve((size_t)N * 4);
    float* den   = carve((size_t)N * 4);
    float* bns   = carve(256);                    // column sums
    float* bnq   = carve(256);                    // column sum-of-squares

    const int heads[4] = {4, 4, 4, 1};
    const int Kin[4]   = {64, 256, 256, 256};

    // feature extractor -> bufA[N,64]
    k_fe<<<(N + 3) / 4, 256, 0, stream>>>(x, fe_w1, fe_b1, fe_w2, fe_b2, bufA, N);

    for (int i = 0; i < 4; ++i) {
        const int hds = heads[i];
        const int K   = Kin[i];
        const int D   = hds * HH;                 // output dim of this layer
        const float* W  = (const float*)d_in[pbase[i] + 0];
        const float* As = (const float*)d_in[pbase[i] + 1];
        const float* Ad = (const float*)d_in[pbase[i] + 2];
        // b{i} skipped: a constant column shift is exactly cancelled by BatchNorm
        const float* g  = (const float*)d_in[pbase[i] + 4];
        const float* be = (const float*)d_in[pbase[i] + 5];

        // 0) operand conversion: A -> bf16 row-major, W -> bf16 transposed
        const long long apairs = (long long)N * K / 2;
        k_cvt_bf16<<<(unsigned)((apairs + 255) / 256), 256, 0, stream>>>(bufA, (uint32_t*)bufAh, apairs);
        k_cvt_w<<<(K * D + 255) / 256, 256, 0, stream>>>(W, Wt, K, D);

        // 1) hT = hin @ W via WMMA
        const int waves = ((N + 15) / 16) * (D / 64);
        k_gemm_bf16<<<(waves + 7) / 8, 256, 0, stream>>>(bufAh, (const __bf16*)Wt, bufB, N, K, D);

        // 2) per-node attention scores
        k_attn<<<(N * hds + 255) / 256, 256, 0, stream>>>(bufB, As, Ad, asrc, adst, N, hds);

        // 3) init accumulators: amax=-inf, den=0, agg=0, bn sums=0
        k_fill_u32<<<(N * hds + 255) / 256, 256, 0, stream>>>((uint32_t*)amax, 0xFF800000u, (long long)N * hds);
        k_fill_u32<<<(N * hds + 255) / 256, 256, 0, stream>>>((uint32_t*)den, 0u, (long long)N * hds);
        const long long nd = (long long)N * D;
        k_fill_u32<<<(unsigned)((nd + 255) / 256), 256, 0, stream>>>((uint32_t*)bufA, 0u, nd);
        k_fill_u32<<<2, 256, 0, stream>>>((uint32_t*)bns, 0u, 512);

        // 4) segment softmax + aggregation over edges (incl. self loops)
        const long long eh = (long long)Etot * hds;
        k_alpha<<<(unsigned)((eh + 255) / 256), 256, 0, stream>>>(src, dst, E, Etot, asrc, adst, alpha, amax, hds);
        k_expsum<<<(unsigned)((eh + 255) / 256), 256, 0, stream>>>(src, dst, E, Etot, alpha, amax, den, hds);
        const long long aggT = eh * 32;
        k_agg<<<(unsigned)((aggT + 255) / 256), 256, 0, stream>>>(src, dst, E, Etot, bufB, alpha, den, bufA, hds);

        // 5) BatchNorm (+ReLU for layers 0..2), in place on bufA
        k_bnstats<<<512, D, 0, stream>>>(bufA, bns, bnq, N, D);
        k_bnapply<<<(unsigned)((nd + 255) / 256), 256, 0, stream>>>(bufA, bns, bnq, g, be, N, D, i < 3 ? 1 : 0);
    }

    // output heads
    k_heads<<<(N + 3) / 4, 256, 0, stream>>>(bufA,
                                             cls_w1, cls_b1, cls_w2, cls_b2,
                                             cnf_w1, cnf_b1, cnf_w2, cnf_b2,
                                             cor_w1, cor_b1, cor_w2, cor_b2,
                                             (float*)d_out, N);
}
